// StableResonantSSM_21096879358598
// MI455X (gfx1250) — compile-verified, matrix operand order checked by reading, not run on hardware
//
#include <hip/hip_runtime.h>
#include <hip/hip_bf16.h>

typedef __attribute__((ext_vector_type(4)))  float    v4f;
typedef __attribute__((ext_vector_type(8)))  float    v8f;
typedef __attribute__((ext_vector_type(16))) _Float16 v16h;

#define HALF_DT 0.005f
#define SQRT_DT 0.1f

static constexpr int BATCH   = 16;
static constexpr int SEQ     = 2048;
static constexpr int D_IN    = 1024;
static constexpr int N_STATE = 128;
static constexpr int M_ROWS  = BATCH * SEQ;   // 32768
static constexpr int N2      = 2 * N_STATE;   // 256 (real | imag columns)
static constexpr int CHUNK   = 128;
static constexpr int NCHUNK  = SEQ / CHUNK;   // 16

// ---------------- workspace layout (bytes) ----------------
static constexpr size_t OFF_ADISC = 0;                                   // 128 * float2
static constexpr size_t OFF_POWK  = 1024;                                // 128*128 * float2 (a^{j+1})
static constexpr size_t OFF_BFRAG = OFF_POWK + 128 * 128 * 8;            // 1024*256 f16 (B_disc fragments)
static constexpr size_t OFF_CFRAG = OFF_BFRAG + 1024 * 256 * 2;          // 128*1024 f16 (C fragments)
static constexpr size_t OFF_LEND  = OFF_CFRAG + 128 * 1024 * 2;          // 16*16*128 float2
static constexpr size_t OFF_HC    = OFF_LEND + 16 * 16 * 128 * 8;        // 16*16*128 float2
static constexpr size_t OFF_BU    = OFF_HC + 16 * 16 * 128 * 8;          // 32768*256 f32
static constexpr size_t OFF_HREAL = OFF_BU + (size_t)M_ROWS * N2 * 4;    // 32768*128 f32

// ---------------- prep: a_disc and power table ----------------
__global__ void k_scalar(const float* __restrict__ Ar, const float* __restrict__ Ai,
                         float2* __restrict__ adisc, float2* __restrict__ powk) {
    int n = threadIdx.x;  // 128 threads
    float ar = Ar[n], ai = Ai[n];
    float nr = 1.f + HALF_DT * ar, ni = HALF_DT * ai;
    float dr = 1.f - HALF_DT * ar, di = -HALF_DT * ai;
    float inv = 1.f / (dr * dr + di * di);
    float adr = (nr * dr + ni * di) * inv;
    float adi = (ni * dr - nr * di) * inv;
    adisc[n] = make_float2(adr, adi);
    float pr = adr, pi = adi;
    for (int j = 0; j < CHUNK; ++j) {       // powk[j] = a^(j+1)
        powk[j * N_STATE + n] = make_float2(pr, pi);
        float t = pr * adr - pi * adi;
        pi = pr * adi + pi * adr;
        pr = t;
    }
}

// ---------------- prep: B_disc packed into WMMA B-fragment layout ----------------
// Bfrag[((kt*16 + nt)*32 + lane)*16 + e], K = d (1024), N = n2 (256: re|im)
__global__ void k_bfrag(const float* __restrict__ B, const float* __restrict__ Ar,
                        const float* __restrict__ Ai, _Float16* __restrict__ Bfrag) {
    int tid  = blockIdx.x * blockDim.x + threadIdx.x;   // 16384 total
    int lane = tid & 31;
    int nt   = (tid >> 5) & 15;
    int kt   = tid >> 9;
    int g    = lane >> 4;
    int n2   = nt * 16 + (lane & 15);
    int n    = n2 & (N_STATE - 1);
    float dr = 1.f - HALF_DT * Ar[n], di = -HALF_DT * Ai[n];
    float inv = 1.f / (dr * dr + di * di);
    float scale = (n2 < N_STATE) ? (SQRT_DT * dr * inv) : (-SQRT_DT * di * inv);
    _Float16* out = Bfrag + (size_t)((kt * 16 + nt) * 32 + lane) * 16;
#pragma unroll
    for (int e = 0; e < 16; ++e) {
        int k = kt * 32 + 8 * g + ((e < 8) ? e : (8 + e));  // ISA 16-bit frag K map
        out[e] = (_Float16)(scale * B[(size_t)n * D_IN + k]);
    }
}

// ---------------- prep: C packed into WMMA B-fragment layout ----------------
// Cfrag[((kt*64 + nt)*32 + lane)*16 + e], K = n (128), N = d (1024)
__global__ void k_cfrag(const float* __restrict__ C, _Float16* __restrict__ Cfrag) {
    int tid  = blockIdx.x * blockDim.x + threadIdx.x;   // 8192 total
    int lane = tid & 31;
    int nt   = (tid >> 5) & 63;
    int kt   = tid >> 11;
    int g    = lane >> 4;
    int d    = nt * 16 + (lane & 15);
    _Float16* out = Cfrag + (size_t)((kt * 64 + nt) * 32 + lane) * 16;
#pragma unroll
    for (int e = 0; e < 16; ++e) {
        int k = kt * 32 + 8 * g + ((e < 8) ? e : (8 + e));
        out[e] = (_Float16)C[(size_t)d * N_STATE + k];
    }
}

// ---------------- GEMM1: Bu[m][n2] = u[m][:] . B_disc^T, f16 WMMA, fp32 accum ----------------
__global__ void __launch_bounds__(256) k_gemm1(const float* __restrict__ u,
                                               const _Float16* __restrict__ Bfrag,
                                               float* __restrict__ Bu) {
    int wave  = threadIdx.x >> 5;
    int lane  = threadIdx.x & 31;
    int g     = lane >> 4;
    int lm    = lane & 15;
    int mtile = blockIdx.x * 2 + (wave >> 2);   // 2048 m-tiles
    int ntb   = (wave & 3) * 4;                 // 4 n-tiles per wave, 16 total
    const float* urow = u + (size_t)(mtile * 16 + lm) * D_IN;
    v8f acc[4] = {};
    for (int kt = 0; kt < 32; ++kt) {
        int k0 = kt * 32 + 8 * g;
        if (kt < 31) __builtin_prefetch(urow + k0 + 32, 0, 1);
        v4f a0 = *(const v4f*)(urow + k0);
        v4f a1 = *(const v4f*)(urow + k0 + 4);
        v4f a2 = *(const v4f*)(urow + k0 + 16);
        v4f a3 = *(const v4f*)(urow + k0 + 20);
        v16h af;
#pragma unroll
        for (int i = 0; i < 4; ++i) {
            af[i]      = (_Float16)a0[i];
            af[4 + i]  = (_Float16)a1[i];
            af[8 + i]  = (_Float16)a2[i];
            af[12 + i] = (_Float16)a3[i];
        }
#pragma unroll
        for (int t = 0; t < 4; ++t) {
            const v16h bf = *(const v16h*)(Bfrag + (size_t)((kt * 16 + ntb + t) * 32 + lane) * 16);
            acc[t] = __builtin_amdgcn_wmma_f32_16x16x32_f16(false, af, false, bf,
                                                            (short)0, acc[t], false, false);
        }
    }
#pragma unroll
    for (int t = 0; t < 4; ++t) {
        int col = (ntb + t) * 16 + lm;
#pragma unroll
        for (int v = 0; v < 8; ++v)
            Bu[(size_t)(mtile * 16 + v + 8 * g) * N2 + col] = acc[t][v];
    }
}

// ---------------- scan pass A: per-chunk local scans (in place over Bu) ----------------
__global__ void k_scanA(float* __restrict__ Bu, const float2* __restrict__ adisc,
                        float2* __restrict__ Lend) {
    int b = blockIdx.x >> 4, c = blockIdx.x & 15, n = threadIdx.x;  // 256 blocks x 128
    float2 a = adisc[n];
    float hr = 0.f, hi = 0.f;
    for (int j = 0; j < CHUNK; ++j) {
        size_t base = (size_t)(b * SEQ + c * CHUNK + j) * N2;
        float xr = Bu[base + n];
        float xi = Bu[base + N_STATE + n];
        float t = a.x * hr - a.y * hi + xr;
        hi = a.x * hi + a.y * hr + xi;
        hr = t;
        Bu[base + n] = hr;
        Bu[base + N_STATE + n] = hi;
    }
    Lend[(b * NCHUNK + c) * N_STATE + n] = make_float2(hr, hi);
}

// ---------------- scan pass B: cross-chunk prefix (multiplier a^128 = powk[127]) ----------------
__global__ void k_scanB(const float2* __restrict__ Lend, const float2* __restrict__ powk,
                        float2* __restrict__ Hc) {
    int b = blockIdx.x, n = threadIdx.x;   // 16 blocks x 128
    float2 aS = powk[(CHUNK - 1) * N_STATE + n];  // a^128
    float Hr = 0.f, Hi = 0.f;
    for (int c = 0; c < NCHUNK; ++c) {
        int idx = (b * NCHUNK + c) * N_STATE + n;
        Hc[idx] = make_float2(Hr, Hi);     // state entering chunk c
        float2 l = Lend[idx];
        float t = aS.x * Hr - aS.y * Hi + l.x;
        Hi = aS.x * Hi + aS.y * Hr + l.y;
        Hr = t;
    }
}

// ---------------- scan pass C: h_t = l_t + a^{j+1} * H_chunk ; emit Re(h) ----------------
__global__ void k_scanC(const float* __restrict__ Bu, const float2* __restrict__ powk,
                        const float2* __restrict__ Hc, float* __restrict__ hreal) {
    int b = blockIdx.x >> 4, c = blockIdx.x & 15, n = threadIdx.x;
    float2 H = Hc[(b * NCHUNK + c) * N_STATE + n];
    for (int j = 0; j < CHUNK; ++j) {
        int m = b * SEQ + c * CHUNK + j;
        float lr = Bu[(size_t)m * N2 + n];
        float2 p = powk[j * N_STATE + n];
        hreal[(size_t)m * N_STATE + n] = lr + p.x * H.x - p.y * H.y;
    }
}

// ---------------- GEMM2: y[m][d] = hreal[m][:] . C^T, f16 WMMA, fp32 accum ----------------
__global__ void __launch_bounds__(256) k_gemm2(const float* __restrict__ hreal,
                                               const _Float16* __restrict__ Cfrag,
                                               float* __restrict__ y) {
    int wave  = threadIdx.x >> 5;
    int lane  = threadIdx.x & 31;
    int g     = lane >> 4;
    int lm    = lane & 15;
    int mtile = blockIdx.x >> 1;                      // 2048 m-tiles
    int ntb   = ((blockIdx.x & 1) * 8 + wave) * 4;    // 64 d-tiles total
    const float* hrow = hreal + (size_t)(mtile * 16 + lm) * N_STATE;
    v8f acc[4] = {};
#pragma unroll
    for (int kt = 0; kt < 4; ++kt) {
        int k0 = kt * 32 + 8 * g;
        v4f a0 = *(const v4f*)(hrow + k0);
        v4f a1 = *(const v4f*)(hrow + k0 + 4);
        v4f a2 = *(const v4f*)(hrow + k0 + 16);
        v4f a3 = *(const v4f*)(hrow + k0 + 20);
        v16h af;
#pragma unroll
        for (int i = 0; i < 4; ++i) {
            af[i]      = (_Float16)a0[i];
            af[4 + i]  = (_Float16)a1[i];
            af[8 + i]  = (_Float16)a2[i];
            af[12 + i] = (_Float16)a3[i];
        }
#pragma unroll
        for (int t = 0; t < 4; ++t) {
            const v16h bf = *(const v16h*)(Cfrag + (size_t)((kt * 64 + ntb + t) * 32 + lane) * 16);
            acc[t] = __builtin_amdgcn_wmma_f32_16x16x32_f16(false, af, false, bf,
                                                            (short)0, acc[t], false, false);
        }
    }
#pragma unroll
    for (int t = 0; t < 4; ++t) {
        int col = (ntb + t) * 16 + lm;
#pragma unroll
        for (int v = 0; v < 8; ++v)
            y[(size_t)(mtile * 16 + v + 8 * g) * D_IN + col] = acc[t][v];
    }
}

extern "C" void kernel_launch(void* const* d_in, const int* in_sizes, int n_in,
                              void* d_out, int out_size, void* d_ws, size_t ws_size,
                              hipStream_t stream) {
    (void)in_sizes; (void)n_in; (void)out_size; (void)ws_size;
    const float* u  = (const float*)d_in[0];
    const float* Ar = (const float*)d_in[1];
    const float* Ai = (const float*)d_in[2];
    const float* B  = (const float*)d_in[3];
    const float* C  = (const float*)d_in[4];
    float* y = (float*)d_out;

    char* ws = (char*)d_ws;
    float2*   adisc = (float2*)(ws + OFF_ADISC);
    float2*   powk  = (float2*)(ws + OFF_POWK);
    _Float16* Bfrag = (_Float16*)(ws + OFF_BFRAG);
    _Float16* Cfrag = (_Float16*)(ws + OFF_CFRAG);
    float2*   Lend  = (float2*)(ws + OFF_LEND);
    float2*   Hc    = (float2*)(ws + OFF_HC);
    float*    Bu    = (float*)(ws + OFF_BU);
    float*    hreal = (float*)(ws + OFF_HREAL);

    k_scalar<<<1, 128, 0, stream>>>(Ar, Ai, adisc, powk);
    k_bfrag <<<64, 256, 0, stream>>>(B, Ar, Ai, Bfrag);
    k_cfrag <<<32, 256, 0, stream>>>(C, Cfrag);
    k_gemm1 <<<M_ROWS / 32, 256, 0, stream>>>(u, Bfrag, Bu);           // 1024 blocks
    k_scanA <<<BATCH * NCHUNK, 128, 0, stream>>>(Bu, adisc, Lend);     // 256 blocks
    k_scanB <<<BATCH, 128, 0, stream>>>(Lend, powk, Hc);
    k_scanC <<<BATCH * NCHUNK, 128, 0, stream>>>(Bu, powk, Hc, hreal);
    k_gemm2 <<<(M_ROWS / 16) * 2, 256, 0, stream>>>(hreal, Cfrag, y);  // 4096 blocks
}
